// GINPolicyNetwork_4329327034728
// MI455X (gfx1250) — matrix operand order repaired and checked by reading, not compiled, not run on hardware
//
#include <hip/hip_runtime.h>
#include <hip/hip_fp16.h>

#define NN 100000
#define EE 1600000
#define GG 256
#define DM 192
#define FF_DIM 2048

typedef __attribute__((ext_vector_type(16))) _Float16 v16h;
typedef __attribute__((ext_vector_type(8)))  _Float16 v8h;
typedef __attribute__((ext_vector_type(8)))  float    v8f;

// ---------------------------------------------------------------------------
// f32 -> f16 convert (for the input features x)
// ---------------------------------------------------------------------------
__global__ __launch_bounds__(256) void f32_to_f16_kernel(
    const float* __restrict__ in, _Float16* __restrict__ out, int n)
{
    int i = blockIdx.x * blockDim.x + threadIdx.x;
    if (i < n) out[i] = (_Float16)in[i];
}

// ---------------------------------------------------------------------------
// Pack a 64x64 f32 weight matrix into the exact WMMA B-register layout:
// out[(jt*2+kh)*512 + lane*16 + hh] = W[j][k],
//   j = jt*16 + (lane&15), k = kh*32 + (hh>=8 ? 16 : 0) + (lane>=16 ? 8 : 0) + (hh&7)
// One contiguous 32B v16h per (jt,kh,lane).
// ---------------------------------------------------------------------------
__global__ __launch_bounds__(256) void pack_weights_kernel(
    const float* __restrict__ W, _Float16* __restrict__ out)
{
    int i = blockIdx.x * blockDim.x + threadIdx.x;   // 0..4095
    if (i >= 4096) return;
    int hh   = i & 15;
    int lane = (i >> 4) & 31;
    int slot = i >> 9;                 // jt*2 + kh
    int jt = slot >> 1, kh = slot & 1;
    int j = jt * 16 + (lane & 15);
    int k = kh * 32 + ((hh >> 3) << 4) + ((lane >> 4) << 3) + (hh & 7);
    out[i] = (_Float16)W[j * 64 + k];
}

// ---------------------------------------------------------------------------
// Edge scatter in f16: agg[dst] += h[src] via packed-f16 atomics
// (global_atomic_pk_add_f16).  One 16B chunk (8 halves) per thread.
// ---------------------------------------------------------------------------
__global__ __launch_bounds__(256) void scatter_add_f16_kernel(
    const _Float16* __restrict__ h,
    const long long* __restrict__ src,
    const long long* __restrict__ dst,
    _Float16* __restrict__ agg)
{
    long long tid = (long long)blockIdx.x * blockDim.x + threadIdx.x;
    if (tid >= (long long)EE * 8) return;
    long long e = tid >> 3;
    int c = (int)(tid & 7);
    long long s = src[e], d = dst[e];

    uint4 u = *(const uint4*)(h + s * 64 + c * 8);        // 8 halves, coalesced
    __half2* ap = (__half2*)(agg + d * 64 + c * 8);
    atomicAdd(ap + 0, __builtin_bit_cast(__half2, u.x));
    atomicAdd(ap + 1, __builtin_bit_cast(__half2, u.y));
    atomicAdd(ap + 2, __builtin_bit_cast(__half2, u.z));
    atomicAdd(ap + 3, __builtin_bit_cast(__half2, u.w));
}

// ---------------------------------------------------------------------------
// Graph pooling: pooled[batch[n]][off + j] += h[n][j]  (f16 -> f32 acc)
// ---------------------------------------------------------------------------
__global__ __launch_bounds__(256) void pool_add_kernel(
    const _Float16* __restrict__ h,
    const long long* __restrict__ batch,
    float* __restrict__ pooled, int off)
{
    int tid = blockIdx.x * blockDim.x + threadIdx.x;
    if (tid >= NN * 64) return;
    int n = tid >> 6, j = tid & 63;
    long long g = batch[n];
    atomicAdd(&pooled[g * DM + off + j], (float)h[tid]);
}

// ---------------------------------------------------------------------------
// Fused GIN MLP (f16 in / f16 out, f32 accumulate):
//   Y = relu( relu(BN(X*W1^T + b1)) * W2^T + b2 )
// One wave per 16-row tile; 16x16x32 WMMA; LDS re-layout between GEMMs;
// packed B-layout weights; coalesced b128 output stores via LDS staging.
// ---------------------------------------------------------------------------
__global__ __launch_bounds__(256) void gin_mlp_wmma_kernel(
    const _Float16* __restrict__ X,
    const _Float16* __restrict__ W1p, const float* __restrict__ B1,
    const float* __restrict__ BNG, const float* __restrict__ BNB,
    const float* __restrict__ BNM, const float* __restrict__ BNV,
    const _Float16* __restrict__ W2p, const float* __restrict__ B2,
    _Float16* __restrict__ Y, int nrows)
{
    __shared__ _Float16 lds1[8][16 * 72];   // padded: GEMM1->GEMM2 staging
    __shared__ _Float16 lds2[8][16 * 64];   // GEMM2 -> packed global stores

    const int lane = threadIdx.x & 31;
    const int wv   = threadIdx.x >> 5;
    int row0 = (blockIdx.x * 8 + wv) * 16;
    if (row0 + 16 > nrows) row0 = nrows - 16;   // clamp (dup work, same values)

    const int mrow = lane & 15;
    const int hi   = (lane >> 4) & 1;           // lane >= 16

    // ---- A tile: ISA A-layout is contiguous K runs per lane half ----
    v16h a0, a1;
    {
        const _Float16* xr = X + (size_t)(row0 + mrow) * 64 + hi * 8;
        v8h x0 = *(const v8h*)(xr);          // K  0.. 7 (+hi*8)
        v8h x1 = *(const v8h*)(xr + 16);     // K 16..23 (+hi*8)
        v8h x2 = *(const v8h*)(xr + 32);     // K 32..39 (+hi*8)
        v8h x3 = *(const v8h*)(xr + 48);     // K 48..55 (+hi*8)
        a0 = __builtin_shufflevector(x0, x1, 0,1,2,3,4,5,6,7,8,9,10,11,12,13,14,15);
        a1 = __builtin_shufflevector(x2, x3, 0,1,2,3,4,5,6,7,8,9,10,11,12,13,14,15);
    }

    // ---- GEMM1 + bias + batchnorm + relu -> LDS1 (f16) ----
    #pragma unroll
    for (int jt = 0; jt < 4; ++jt) {
        int col = jt * 16 + mrow;
        v16h b0 = *(const v16h*)(W1p + (jt * 2 + 0) * 512 + lane * 16);
        v16h b1 = *(const v16h*)(W1p + (jt * 2 + 1) * 512 + lane * 16);
        v8f acc = {};
        acc = __builtin_amdgcn_wmma_f32_16x16x32_f16(false, a0, false, b0,
                                                     (short)0, acc, false, false);
        acc = __builtin_amdgcn_wmma_f32_16x16x32_f16(false, a1, false, b1,
                                                     (short)0, acc, false, false);
        float sc = BNG[col] * rsqrtf(BNV[col] + 1e-5f);
        float sh = (B1[col] - BNM[col]) * sc + BNB[col];
        #pragma unroll
        for (int r = 0; r < 8; ++r) {
            int mr = hi * 8 + r;
            float val = acc[r] * sc + sh;
            val = val > 0.f ? val : 0.f;
            lds1[wv][mr * 72 + col] = (_Float16)val;
        }
    }

    asm volatile("s_wait_dscnt 0" ::: "memory");   // per-wave LDS RAW fence

    // ---- re-load intermediate as A tile for GEMM2 ----
    v16h c0, c1;
    {
        const _Float16* lr = &lds1[wv][mrow * 72 + hi * 8];
        v8h x0 = *(const v8h*)(lr);
        v8h x1 = *(const v8h*)(lr + 16);
        v8h x2 = *(const v8h*)(lr + 32);
        v8h x3 = *(const v8h*)(lr + 48);
        c0 = __builtin_shufflevector(x0, x1, 0,1,2,3,4,5,6,7,8,9,10,11,12,13,14,15);
        c1 = __builtin_shufflevector(x2, x3, 0,1,2,3,4,5,6,7,8,9,10,11,12,13,14,15);
    }

    // ---- GEMM2 + bias + relu -> LDS2 (f16, conflict-free row-major) ----
    #pragma unroll
    for (int jt = 0; jt < 4; ++jt) {
        int col = jt * 16 + mrow;
        v16h b0 = *(const v16h*)(W2p + (jt * 2 + 0) * 512 + lane * 16);
        v16h b1 = *(const v16h*)(W2p + (jt * 2 + 1) * 512 + lane * 16);
        v8f acc = {};
        acc = __builtin_amdgcn_wmma_f32_16x16x32_f16(false, c0, false, b0,
                                                     (short)0, acc, false, false);
        acc = __builtin_amdgcn_wmma_f32_16x16x32_f16(false, c1, false, b1,
                                                     (short)0, acc, false, false);
        float bias = B2[col];
        #pragma unroll
        for (int r = 0; r < 8; ++r) {
            int mr = hi * 8 + r;
            float val = acc[r] + bias;
            lds2[wv][mr * 64 + col] = (_Float16)(val > 0.f ? val : 0.f);
        }
    }

    asm volatile("s_wait_dscnt 0" ::: "memory");

    // ---- packed store: tile is contiguous in memory (full 64-wide rows) ----
    {
        const _Float16* l2 = &lds2[wv][lane * 32];
        _Float16* yr = Y + (size_t)row0 * 64 + lane * 32;
        *(v8h*)(yr)      = *(const v8h*)(l2);
        *(v8h*)(yr + 8)  = *(const v8h*)(l2 + 8);
        *(v8h*)(yr + 16) = *(const v8h*)(l2 + 16);
        *(v8h*)(yr + 24) = *(const v8h*)(l2 + 24);
    }
}

// ---------------------------------------------------------------------------
// Stage B1: seq-len==1 attention degenerates to attn==V.
// out = LayerNorm(pooled + (pooled*Wv^T + bv)*Wo^T + bo; ln1)
// ---------------------------------------------------------------------------
__global__ __launch_bounds__(256) void attn_ln1_kernel(
    const float* __restrict__ pooled,
    const float* __restrict__ qkv_w, const float* __restrict__ qkv_b,
    const float* __restrict__ wo, const float* __restrict__ bo,
    const float* __restrict__ g1, const float* __restrict__ b1,
    float* __restrict__ out)
{
    const int g = blockIdx.x, t = threadIdx.x;
    __shared__ float p[DM], vv[DM], red[256];

    if (t < DM) p[t] = pooled[g * DM + t];
    __syncthreads();

    if (t < DM) {                                   // V projection only
        const float* wr = qkv_w + (size_t)(2 * DM + t) * DM;
        float s = qkv_b[2 * DM + t];
        for (int k = 0; k < DM; ++k) s += p[k] * wr[k];
        vv[t] = s;
    }
    __syncthreads();

    float a = 0.f;
    if (t < DM) {                                   // output projection + residual
        const float* wr = wo + (size_t)t * DM;
        float s = bo[t];
        for (int k = 0; k < DM; ++k) s += vv[k] * wr[k];
        a = p[t] + s;
    }

    red[t] = (t < DM) ? a : 0.f; __syncthreads();
    for (int off = 128; off > 0; off >>= 1) { if (t < off) red[t] += red[t + off]; __syncthreads(); }
    float mu = red[0] / (float)DM; __syncthreads();
    float d = (t < DM) ? (a - mu) : 0.f;
    red[t] = d * d; __syncthreads();
    for (int off = 128; off > 0; off >>= 1) { if (t < off) red[t] += red[t + off]; __syncthreads(); }
    float rs = rsqrtf(red[0] / (float)DM + 1e-5f);
    if (t < DM) out[g * DM + t] = d * rs * g1[t] + b1[t];
}

// ---------------------------------------------------------------------------
// Stage B2: FF + LN2 + head.  One block per graph.
// ---------------------------------------------------------------------------
__global__ __launch_bounds__(256) void ff_ln2_head_kernel(
    const float* __restrict__ hin,
    const float* __restrict__ ff1w, const float* __restrict__ ff1b,
    const float* __restrict__ ff2w, const float* __restrict__ ff2b,
    const float* __restrict__ g2, const float* __restrict__ b2,
    const float* __restrict__ l1w, const float* __restrict__ l1b,
    const float* __restrict__ l2w, const float* __restrict__ l2b,
    float* __restrict__ out)
{
    const int g = blockIdx.x, t = threadIdx.x;
    __shared__ float h[DM], f[FF_DIM], h2[DM], r[DM], red[256];

    if (t < DM) h[t] = hin[g * DM + t];
    __syncthreads();

    for (int m = t; m < FF_DIM; m += 256) {
        const float* wr = ff1w + (size_t)m * DM;
        float s = ff1b[m];
        for (int k = 0; k < DM; ++k) s += h[k] * wr[k];
        f[m] = s > 0.f ? s : 0.f;
    }
    __syncthreads();

    float a = 0.f;
    if (t < DM) {
        const float* wr = ff2w + (size_t)t * FF_DIM;
        float s = ff2b[t];
        for (int k = 0; k < FF_DIM; ++k) s += f[k] * wr[k];
        a = h[t] + s;
    }

    red[t] = (t < DM) ? a : 0.f; __syncthreads();
    for (int off = 128; off > 0; off >>= 1) { if (t < off) red[t] += red[t + off]; __syncthreads(); }
    float mu = red[0] / (float)DM; __syncthreads();
    float d = (t < DM) ? (a - mu) : 0.f;
    red[t] = d * d; __syncthreads();
    for (int off = 128; off > 0; off >>= 1) { if (t < off) red[t] += red[t + off]; __syncthreads(); }
    float rs = rsqrtf(red[0] / (float)DM + 1e-5f);
    if (t < DM) h2[t] = d * rs * g2[t] + b2[t];
    __syncthreads();

    if (t < DM) {
        const float* wr = l1w + (size_t)t * DM;
        float s = l1b[t];
        for (int k = 0; k < DM; ++k) s += h2[k] * wr[k];
        r[t] = s > 0.f ? s : 0.f;
    }
    __syncthreads();

    red[t] = (t < DM) ? r[t] * l2w[t] : 0.f; __syncthreads();
    for (int off = 128; off > 0; off >>= 1) { if (t < off) red[t] += red[t + off]; __syncthreads(); }
    if (t == 0) out[g] = red[0] + l2b[0];
}

// ---------------------------------------------------------------------------
extern "C" void kernel_launch(void* const* d_in, const int* in_sizes, int n_in,
                              void* d_out, int out_size, void* d_ws, size_t ws_size,
                              hipStream_t stream)
{
    const float*     x     = (const float*)d_in[0];
    const long long* ei    = (const long long*)d_in[1];
    const long long* batch = (const long long*)d_in[2];
    const long long* src   = ei;
    const long long* dst   = ei + EE;

    const float* nn1_w1 = (const float*)d_in[3];
    const float* nn1_b1 = (const float*)d_in[4];
    const float* nn1_g  = (const float*)d_in[5];
    const float* nn1_bb = (const float*)d_in[6];
    const float* nn1_m  = (const float*)d_in[7];
    const float* nn1_v  = (const float*)d_in[8];
    const float* nn1_w2 = (const float*)d_in[9];
    const float* nn1_b2 = (const float*)d_in[10];
    const float* nn2_w1 = (const float*)d_in[11];
    const float* nn2_b1 = (const float*)d_in[12];
    const float* nn2_g  = (const float*)d_in[13];
    const float* nn2_bb = (const float*)d_in[14];
    const float* nn2_m  = (const float*)d_in[15];
    const float* nn2_v  = (const float*)d_in[16];
    const float* nn2_w2 = (const float*)d_in[17];
    const float* nn2_b2 = (const float*)d_in[18];
    const float* qkv_w  = (const float*)d_in[19];
    const float* qkv_b  = (const float*)d_in[20];
    const float* ao_w   = (const float*)d_in[21];
    const float* ao_b   = (const float*)d_in[22];
    const float* ln1_g  = (const float*)d_in[23];
    const float* ln1_b  = (const float*)d_in[24];
    const float* ff1_w  = (const float*)d_in[25];
    const float* ff1_b  = (const float*)d_in[26];
    const float* ff2_w  = (const float*)d_in[27];
    const float* ff2_b  = (const float*)d_in[28];
    const float* ln2_g  = (const float*)d_in[29];
    const float* ln2_b  = (const float*)d_in[30];
    const float* lin1_w = (const float*)d_in[31];
    const float* lin1_b = (const float*)d_in[32];
    const float* lin2_w = (const float*)d_in[33];
    const float* lin2_b = (const float*)d_in[34];

    const size_t N64 = (size_t)NN * 64;

    // workspace layout: f32 region first, then f16 region
    float* pooled = (float*)d_ws;                       // GG*DM
    float* hln1   = pooled + (size_t)GG * DM;           // GG*DM
    _Float16* xh  = (_Float16*)(hln1 + (size_t)GG * DM);
    _Float16* agg = xh  + N64;
    _Float16* h1  = agg + N64;
    _Float16* h2  = h1  + N64;
    _Float16* h3  = h2  + N64;
    _Float16* w11p = h3   + N64;        // 4096 each, WMMA-B packed
    _Float16* w12p = w11p + 4096;
    _Float16* w21p = w12p + 4096;
    _Float16* w22p = w21p + 4096;

    const size_t hbytes = N64 * sizeof(_Float16);
    const int cvt_blocks     = (int)((N64 + 255) / 256);
    const int scatter_blocks = (int)(((long long)EE * 8 + 255) / 256);
    const int mlp_blocks     = (NN / 16 + 7) / 8;       // 8 waves per block
    const int pool_blocks    = (int)((N64 + 255) / 256);

    // ---- prep: x -> f16, pack all weights into WMMA B layout ----
    f32_to_f16_kernel<<<cvt_blocks, 256, 0, stream>>>(x, xh, (int)N64);
    pack_weights_kernel<<<16, 256, 0, stream>>>(nn1_w1, w11p);
    pack_weights_kernel<<<16, 256, 0, stream>>>(nn1_w2, w12p);
    pack_weights_kernel<<<16, 256, 0, stream>>>(nn2_w1, w21p);
    pack_weights_kernel<<<16, 256, 0, stream>>>(nn2_w2, w22p);

    // ---- GIN layer 1 ----
    hipMemcpyAsync(agg, xh, hbytes, hipMemcpyDeviceToDevice, stream);
    scatter_add_f16_kernel<<<scatter_blocks, 256, 0, stream>>>(xh, src, dst, agg);
    gin_mlp_wmma_kernel<<<mlp_blocks, 256, 0, stream>>>(
        agg, w11p, nn1_b1, nn1_g, nn1_bb, nn1_m, nn1_v, w12p, nn1_b2, h1, NN);

    // ---- GIN layer 2 ----
    hipMemcpyAsync(agg, h1, hbytes, hipMemcpyDeviceToDevice, stream);
    scatter_add_f16_kernel<<<scatter_blocks, 256, 0, stream>>>(h1, src, dst, agg);
    gin_mlp_wmma_kernel<<<mlp_blocks, 256, 0, stream>>>(
        agg, w21p, nn2_b1, nn2_g, nn2_bb, nn2_m, nn2_v, w22p, nn2_b2, h2, NN);

    // ---- GIN layer 3 ----
    hipMemcpyAsync(agg, h2, hbytes, hipMemcpyDeviceToDevice, stream);
    scatter_add_f16_kernel<<<scatter_blocks, 256, 0, stream>>>(h2, src, dst, agg);
    gin_mlp_wmma_kernel<<<mlp_blocks, 256, 0, stream>>>(
        agg, w21p, nn2_b1, nn2_g, nn2_bb, nn2_m, nn2_v, w22p, nn2_b2, h3, NN);

    // ---- pooling ----
    hipMemsetAsync(pooled, 0, (size_t)GG * DM * sizeof(float), stream);
    pool_add_kernel<<<pool_blocks, 256, 0, stream>>>(h1, batch, pooled, 0);
    pool_add_kernel<<<pool_blocks, 256, 0, stream>>>(h2, batch, pooled, 64);
    pool_add_kernel<<<pool_blocks, 256, 0, stream>>>(h3, batch, pooled, 128);

    // ---- transformer (seq len 1 -> attn == V) + head ----
    attn_ln1_kernel<<<GG, 256, 0, stream>>>(pooled, qkv_w, qkv_b, ao_w, ao_b,
                                            ln1_g, ln1_b, hln1);
    ff_ln2_head_kernel<<<GG, 256, 0, stream>>>(hln1, ff1_w, ff1_b, ff2_w, ff2_b,
                                               ln2_g, ln2_b, lin1_w, lin1_b,
                                               lin2_w, lin2_b, (float*)d_out);
}
